// LinearBit_69114613729843
// MI455X (gfx1250) — matrix operand order repaired and checked by reading, not compiled, not run on hardware
//
#include <hip/hip_runtime.h>
#include <hip/hip_bf16.h>

typedef __attribute__((ext_vector_type(16))) __bf16 bf16x16;
typedef __attribute__((ext_vector_type(8)))  __bf16 bf16x8;
typedef __attribute__((ext_vector_type(8)))  float  f32x8;
typedef __attribute__((ext_vector_type(4)))  float  f32x4;

#define M_DIM 8192
#define K_DIM 4096
#define N_DIM 4096
#define PLANE (K_DIM * N_DIM)   // 16777216

// ---------------------------------------------------------------------------
// Kernel 1: decode bit-plane weights -> bf16 wT[n][k] (transposed), and
// per-block partial sum / sum-of-squares for the std() computation.
// ---------------------------------------------------------------------------
__global__ __launch_bounds__(256) void quant_w_kernel(
    const float* __restrict__ mag, const float* __restrict__ sgn,
    __bf16* __restrict__ wT, float* __restrict__ psum, float* __restrict__ psq)
{
    __shared__ __bf16 tile[64 * 72];
    __shared__ float reds[256];
    __shared__ float redq[256];

    const int t  = threadIdx.x;
    const int n0 = blockIdx.x * 64;
    const int k0 = blockIdx.y * 64;

    float lsum = 0.f, lsq = 0.f;
#pragma unroll
    for (int i = 0; i < 16; ++i) {
        int lin = i * 256 + t;
        int kl = lin >> 6;            // 0..63
        int nl = lin & 63;            // coalesced along n
        int idx = (k0 + kl) * N_DIM + (n0 + nl);
        float m0v = __builtin_nontemporal_load(&mag[idx]);
        float m1v = __builtin_nontemporal_load(&mag[PLANE + idx]);
        float m2v = __builtin_nontemporal_load(&mag[2 * PLANE + idx]);
        float sv  = __builtin_nontemporal_load(&sgn[idx]);
        float b0 = (m0v >= 0.f) ? 0.5f   : 0.f;
        float b1 = (m1v >= 0.f) ? 0.25f  : 0.f;
        float b2 = (m2v >= 0.f) ? 0.125f : 0.f;
        float s  = (sv  >= 0.f) ? 1.f : -1.f;
        float v  = s * (b0 + b1 + b2);
        lsum += v;
        lsq  += v * v;
        tile[nl * 72 + kl] = (__bf16)v;   // exact: <=3 mantissa bits
    }
    reds[t] = lsum;
    redq[t] = lsq;
    __syncthreads();

    // transposed write-out, coalesced along k (re-read by GEMM: regular store)
#pragma unroll
    for (int i = 0; i < 16; ++i) {
        int lin = i * 256 + t;
        int nl = lin >> 6;
        int kl = lin & 63;
        wT[(size_t)(n0 + nl) * K_DIM + (k0 + kl)] = tile[nl * 72 + kl];
    }

    for (int s = 128; s > 0; s >>= 1) {
        if (t < s) { reds[t] += reds[t + s]; redq[t] += redq[t + s]; }
        __syncthreads();
    }
    if (t == 0) {
        int b = blockIdx.y * gridDim.x + blockIdx.x;
        psum[b] = reds[0];
        psq[b]  = redq[0];
    }
}

// ---------------------------------------------------------------------------
// Kernel 2: final reduce of 4096 partials -> alpha scalar
// ---------------------------------------------------------------------------
__global__ __launch_bounds__(256) void alpha_kernel(
    const float* __restrict__ psum, const float* __restrict__ psq,
    float* __restrict__ alpha)
{
    __shared__ float reds[256];
    __shared__ float redq[256];
    const int t = threadIdx.x;
    float s = 0.f, q = 0.f;
    for (int i = t; i < 4096; i += 256) { s += psum[i]; q += psq[i]; }
    reds[t] = s;
    redq[t] = q;
    __syncthreads();
    for (int st = 128; st > 0; st >>= 1) {
        if (t < st) { reds[t] += reds[t + st]; redq[t] += redq[t + st]; }
        __syncthreads();
    }
    if (t == 0) {
        const float ntot = (float)PLANE;
        float mean = reds[0] / ntot;
        float var  = redq[0] / ntot - mean * mean;
        var = var > 0.f ? var : 0.f;
        float stdv = sqrtf(var);
        const float target = 0.022097086912079608f;  // sqrt(2/4096)
        alpha[0] = target / (stdv + 1e-8f);
    }
}

// ---------------------------------------------------------------------------
// Kernel 3: x f32 -> bf16 (8 elements / thread, 128-bit traffic)
// ---------------------------------------------------------------------------
__global__ __launch_bounds__(256) void xcvt_kernel(
    const float* __restrict__ x, __bf16* __restrict__ xb)
{
    size_t gid = (size_t)blockIdx.x * 256 + threadIdx.x;
    const f32x4* xv = (const f32x4*)x;
    f32x4 a = __builtin_nontemporal_load(&xv[gid * 2]);
    f32x4 b = __builtin_nontemporal_load(&xv[gid * 2 + 1]);
    bf16x8 o;
    o[0] = (__bf16)a.x; o[1] = (__bf16)a.y; o[2] = (__bf16)a.z; o[3] = (__bf16)a.w;
    o[4] = (__bf16)b.x; o[5] = (__bf16)b.y; o[6] = (__bf16)b.z; o[7] = (__bf16)b.w;
    ((bf16x8*)xb)[gid] = o;
}

// ---------------------------------------------------------------------------
// Kernel 4: bf16 WMMA GEMM with ping-pong LDS (108 KB dynamic).
// Block tile 128x256x64, 8 waves (2x4), wave tile 64x64 -> 32 WMMAs/stage.
// One barrier per K-stage: compute from buf, prefetch regs -> other buf.
// ---------------------------------------------------------------------------
#define GBM 128
#define GBN 256
#define GBK 64
#define GBKP 72
#define TILE_ELEMS ((GBM + GBN) * GBKP)   // per-buffer __bf16 elements (27648)
#define GEMM_LDS_BYTES (2 * TILE_ELEMS * 2)  // 110,592 B

struct GemmCtx {
    int ldrow, ldcol, wm, wn, h, mr, m0, n0;
};

__device__ __forceinline__ void load_stage_regs(
    const __bf16* __restrict__ A, const __bf16* __restrict__ Bt,
    const GemmCtx& c, int kpos, bf16x8 pa[4], bf16x8 pb[8])
{
#pragma unroll
    for (int p = 0; p < 4; ++p)
        pa[p] = *(const bf16x8*)&A[(size_t)(c.m0 + c.ldrow + p * 32) * K_DIM + kpos + c.ldcol];
#pragma unroll
    for (int p = 0; p < 8; ++p)
        pb[p] = *(const bf16x8*)&Bt[(size_t)(c.n0 + c.ldrow + p * 32) * K_DIM + kpos + c.ldcol];
}

__device__ __forceinline__ void commit_stage_lds(
    __bf16* As, __bf16* Bs, const GemmCtx& c,
    const bf16x8 pa[4], const bf16x8 pb[8])
{
#pragma unroll
    for (int p = 0; p < 4; ++p)
        *(bf16x8*)&As[(c.ldrow + p * 32) * GBKP + c.ldcol] = pa[p];
#pragma unroll
    for (int p = 0; p < 8; ++p)
        *(bf16x8*)&Bs[(c.ldrow + p * 32) * GBKP + c.ldcol] = pb[p];
}

__device__ __forceinline__ void compute_stage(
    const __bf16* As, const __bf16* Bs, const GemmCtx& c, f32x8 acc[4][4])
{
#pragma unroll
    for (int kk = 0; kk < GBK; kk += 32) {
        bf16x16 afr[4];
        bf16x16 bfr[4];
        // A fragment 16x32: lane holds row mr, K chunks {8h..8h+7, 16+8h..+7}
#pragma unroll
        for (int mi = 0; mi < 4; ++mi) {
            const __bf16* base = &As[(c.wm * 64 + mi * 16 + c.mr) * GBKP + kk + 8 * c.h];
            bf16x8 lo = *(const bf16x8*)base;
            bf16x8 hi = *(const bf16x8*)(base + 16);
            afr[mi] = __builtin_shufflevector(lo, hi,
                0, 1, 2, 3, 4, 5, 6, 7, 8, 9, 10, 11, 12, 13, 14, 15);
        }
        // B fragment 32x16: lane holds column mr, K chunks {8h..8h+7, 16+8h..+7}
#pragma unroll
        for (int ni = 0; ni < 4; ++ni) {
            const __bf16* base = &Bs[(c.wn * 64 + ni * 16 + c.mr) * GBKP + kk + 8 * c.h];
            bf16x8 lo = *(const bf16x8*)base;
            bf16x8 hi = *(const bf16x8*)(base + 16);
            bfr[ni] = __builtin_shufflevector(lo, hi,
                0, 1, 2, 3, 4, 5, 6, 7, 8, 9, 10, 11, 12, 13, 14, 15);
        }
#pragma unroll
        for (int mi = 0; mi < 4; ++mi)
#pragma unroll
            for (int ni = 0; ni < 4; ++ni)
                acc[mi][ni] = __builtin_amdgcn_wmma_f32_16x16x32_bf16(
                    false, afr[mi], false, bfr[ni],
                    (short)0, acc[mi][ni], false, false);
    }
}

__global__ __launch_bounds__(256) void gemm_kernel(
    const __bf16* __restrict__ A,   // [M][K] row-major bf16
    const __bf16* __restrict__ Bt,  // [N][K] row-major bf16 (B transposed)
    float* __restrict__ C,          // [M][N]
    const float* __restrict__ alpha_p)
{
    extern __shared__ __align__(16) char smem_raw[];
    __bf16* As0 = (__bf16*)smem_raw;
    __bf16* Bs0 = As0 + GBM * GBKP;
    __bf16* As1 = As0 + TILE_ELEMS;
    __bf16* Bs1 = As1 + GBM * GBKP;

    const int t    = threadIdx.x;
    const int lane = t & 31;
    const int wave = t >> 5;

    GemmCtx c;
    c.wm    = wave >> 2;           // 0..1 -> 64-row slab
    c.wn    = wave & 3;            // 0..3 -> 64-col slab
    c.m0    = blockIdx.y * GBM;
    c.n0    = blockIdx.x * GBN;
    c.h     = lane >> 4;
    c.mr    = lane & 15;
    c.ldrow = t >> 3;              // 0..31
    c.ldcol = (t & 7) * 8;         // 0..56

    f32x8 acc[4][4] = {};

    // ---- prologue: stage K-stage 0 into buffer 0 ----
    {
        bf16x8 pa[4], pb[8];
        load_stage_regs(A, Bt, c, 0, pa, pb);
        commit_stage_lds(As0, Bs0, c, pa, pb);
    }
    __syncthreads();

    // ---- ping-pong main loop: two K-stages per iteration ----
    for (int k0 = 0; k0 < K_DIM; k0 += 2 * GBK) {
        // stage A: compute from buf0, prefetch k0+GBK into buf1
        {
            bf16x8 pa[4], pb[8];
            const bool nxt = (k0 + GBK) < K_DIM;
            if (nxt) load_stage_regs(A, Bt, c, k0 + GBK, pa, pb);
            compute_stage(As0, Bs0, c, acc);
            if (nxt) commit_stage_lds(As1, Bs1, c, pa, pb);
            __syncthreads();
        }
        // stage B: compute from buf1, prefetch k0+2*GBK into buf0
        if ((k0 + GBK) < K_DIM) {
            bf16x8 pa[4], pb[8];
            const bool nxt = (k0 + 2 * GBK) < K_DIM;
            if (nxt) load_stage_regs(A, Bt, c, k0 + 2 * GBK, pa, pb);
            compute_stage(As1, Bs1, c, acc);
            if (nxt) commit_stage_lds(As0, Bs0, c, pa, pb);
            __syncthreads();
        }
    }

    const float alpha = alpha_p[0];
    // D layout: lanes 0-15 -> N=lane, M=r; lanes 16-31 -> N=lane-16, M=8+r
#pragma unroll
    for (int mi = 0; mi < 4; ++mi)
#pragma unroll
        for (int ni = 0; ni < 4; ++ni)
#pragma unroll
            for (int r = 0; r < 8; ++r) {
                int row = c.m0 + c.wm * 64 + mi * 16 + 8 * c.h + r;
                int col = c.n0 + c.wn * 64 + ni * 16 + c.mr;
                __builtin_nontemporal_store(acc[mi][ni][r] * alpha,
                                            &C[(size_t)row * N_DIM + col]);
            }
}

// ---------------------------------------------------------------------------
// Host-side launcher
// ---------------------------------------------------------------------------
extern "C" void kernel_launch(void* const* d_in, const int* in_sizes, int n_in,
                              void* d_out, int out_size, void* d_ws, size_t ws_size,
                              hipStream_t stream) {
    const float* x   = (const float*)d_in[0];  // [8192,4096]
    const float* mag = (const float*)d_in[1];  // [3,4096,4096]
    const float* sgn = (const float*)d_in[2];  // [4096,4096]
    float* out = (float*)d_out;                // [8192,4096]

    char* ws = (char*)d_ws;
    const size_t ALPHA_OFF = 0;                 // 1 float
    const size_t PSUM_OFF  = 1024;              // 4096 floats
    const size_t PSQ_OFF   = PSUM_OFF + 4096 * sizeof(float);
    const size_t XB_OFF    = 65536;             // 64 MB of bf16 x
    const size_t WT_OFF    = XB_OFF + (size_t)M_DIM * K_DIM * sizeof(__bf16);

    float*  alpha = (float*)(ws + ALPHA_OFF);
    float*  psum  = (float*)(ws + PSUM_OFF);
    float*  psq   = (float*)(ws + PSQ_OFF);
    __bf16* xb    = (__bf16*)(ws + XB_OFF);
    __bf16* wT    = (__bf16*)(ws + WT_OFF);

    // >64KB dynamic LDS opt-in (CDNA5 allows up to 320KB per workgroup).
    (void)hipFuncSetAttribute((const void*)gemm_kernel,
                              hipFuncAttributeMaxDynamicSharedMemorySize,
                              GEMM_LDS_BYTES);

    quant_w_kernel<<<dim3(N_DIM / 64, K_DIM / 64), 256, 0, stream>>>(
        mag, sgn, wT, psum, psq);
    alpha_kernel<<<1, 256, 0, stream>>>(psum, psq, alpha);
    xcvt_kernel<<<(int)(((size_t)M_DIM * K_DIM) / (256 * 8)), 256, 0, stream>>>(x, xb);
    gemm_kernel<<<dim3(N_DIM / 256, M_DIM / 128), 256, GEMM_LDS_BYTES, stream>>>(
        xb, wT, out, alpha);
}
